// CrossAttention_32650341384855
// MI455X (gfx1250) — matrix operand order repaired and checked by reading, not compiled, not run on hardware
//
#include <hip/hip_runtime.h>
#include <hip/hip_bf16.h>

#define BB 4
#define NN 1024
#define CC 1024
#define EE 1024
#define HH 16
#define DD 64

typedef __attribute__((ext_vector_type(16))) __bf16 v16bf;
typedef __attribute__((ext_vector_type(8)))  float  v8f;
typedef __attribute__((ext_vector_type(4)))  int    v4i;

#define AS1 __attribute__((address_space(1)))
#define AS3 __attribute__((address_space(3)))

#if __has_builtin(__builtin_amdgcn_global_load_async_to_lds_b128) && \
    __has_builtin(__builtin_amdgcn_s_wait_asynccnt)
#define HAVE_ASYNC 1
#else
#define HAVE_ASYNC 0
#endif

union BFrag {
    v16bf v;
    unsigned short u[16];
};

__device__ __forceinline__ unsigned short f2bf(float f) {
    unsigned int u = __float_as_uint(f);
    u += 0x7FFFu + ((u >> 16) & 1u);   // round-to-nearest-even
    return (unsigned short)(u >> 16);
}

__device__ __forceinline__ unsigned int f2bf2(float lo, float hi) {
#if __has_builtin(__builtin_amdgcn_cvt_pk_bf16_f32)
    typedef __attribute__((ext_vector_type(2))) __bf16 v2bf;
    v2bf t = __builtin_amdgcn_cvt_pk_bf16_f32(lo, hi);
    return __builtin_bit_cast(unsigned int, t);
#else
    return (unsigned int)f2bf(lo) | ((unsigned int)f2bf(hi) << 16);
#endif
}

// async 16-byte global->LDS copy (falls back to sync copy through VGPRs)
__device__ __forceinline__ void cp16(const void* g, void* l) {
#if HAVE_ASYNC
    __builtin_amdgcn_global_load_async_to_lds_b128(
        (AS1 v4i*)(unsigned long long)g, (AS3 v4i*)(unsigned long long)l, 0, 0);
#else
    *(uint4*)l = *(const uint4*)g;
#endif
}

__device__ __forceinline__ void cp_wait() {
#if HAVE_ASYNC
    __builtin_amdgcn_s_wait_asynccnt(0);
#endif
}

// ---------------------------------------------------------------------------
// Kernel 0a: flat fp32 -> bf16 convert (x, query), 8 elements/thread
// ---------------------------------------------------------------------------
__global__ __launch_bounds__(256) void cvt_bf16_kernel(
    const float* __restrict__ src, unsigned short* __restrict__ dst, int n) {
    int i = (blockIdx.x * 256 + threadIdx.x) * 8;
    if (i < n) {
        float4 a = *(const float4*)&src[i];
        float4 b = *(const float4*)&src[i + 4];
        uint4 o;
        o.x = f2bf2(a.x, a.y);
        o.y = f2bf2(a.z, a.w);
        o.z = f2bf2(b.x, b.y);
        o.w = f2bf2(b.z, b.w);
        *(uint4*)&dst[i] = o;
    }
}

// ---------------------------------------------------------------------------
// Kernel 0b: transpose + convert weights: src (K x COLS f32) -> dst (COLS x K bf16)
// ---------------------------------------------------------------------------
__global__ __launch_bounds__(256) void transpose_bf16_kernel(
    const float* __restrict__ src, unsigned short* __restrict__ dst,
    int K, int COLS) {
    __shared__ float tile[32][33];
    const int tx = threadIdx.x & 31;
    const int ty = threadIdx.x >> 5;        // 0..7
    const int col0 = blockIdx.x * 32;
    const int k0 = blockIdx.y * 32;
#pragma unroll
    for (int i = 0; i < 32; i += 8)
        tile[ty + i][tx] = src[(size_t)(k0 + ty + i) * COLS + col0 + tx];
    __syncthreads();
#pragma unroll
    for (int i = 0; i < 32; i += 8)
        dst[(size_t)(col0 + ty + i) * K + k0 + tx] = f2bf(tile[tx][ty + i]);
}

// ---------------------------------------------------------------------------
// Kernel 1: fused projections on bf16 data, double-buffered async staging.
// Combined columns 3072: [0,1024)->K (row major), [1024,2048)->V (stored
// TRANSPOSED per (b,h): VbufT[((b*H+h)*D+d)*N + n]), [2048,3072)->Q.
// Block = 8 waves; block tile 64 rows x 128 cols; wave = 64x16 (4 WMMA/kstep).
// ---------------------------------------------------------------------------
__global__ __launch_bounds__(256) void proj_kernel(
    const unsigned short* __restrict__ xb, const unsigned short* __restrict__ qb,
    const unsigned short* __restrict__ WkvT, const unsigned short* __restrict__ WqT,
    const float* __restrict__ bkv, const float* __restrict__ bq,
    unsigned short* __restrict__ Kbuf, unsigned short* __restrict__ VbufT,
    unsigned short* __restrict__ Qbuf) {
    __shared__ unsigned short lA[2][64 * 40];   // 64 rows x 32 k, stride 40
    __shared__ unsigned short lW[2][128 * 40];  // 128 cols x 32 k, stride 40

    const int tid = threadIdx.x;
    const int lane = tid & 31;
    const int wave = tid >> 5;
    const int rowBase = blockIdx.x * 64;      // 64 blocks over B*N=4096 rows
    const int colBase = blockIdx.y * 128;     // 24 blocks over 3072 cols
    const bool isQ = (colBase >= 2 * EE);
    const unsigned short* Asrc = isQ ? qb : xb;
    const unsigned short* WT = isQ ? WqT : WkvT;
    const int wcolBase = isQ ? (colBase - 2 * EE) : colBase;

    const int kgrp = (lane >> 4) * 8;          // A-frag K group
    const int kbase = (lane >> 4) * 8;         // B-frag K base
    const int r0 = lane & 15;
    const int wcol = wave * 16 + (lane & 15);

    // stage one 32-wide k-slice into buffer `buf`
    auto stage = [&](int kk, int buf) {
        {   // A: 64 rows x 64B, 256 x 16B transfers
            int r = tid >> 2, s = (tid & 3) * 8;
            cp16(&Asrc[(size_t)(rowBase + r) * CC + kk + s], &lA[buf][r * 40 + s]);
        }
        {   // W: 128 cols x 64B, 512 x 16B transfers (2/thread)
#pragma unroll
            for (int q = 0; q < 2; ++q) {
                int id = tid * 2 + q;
                int c = id >> 2, s = (id & 3) * 8;
                cp16(&WT[(size_t)(wcolBase + c) * CC + kk + s], &lW[buf][c * 40 + s]);
            }
        }
    };

    v8f acc[4] = {{}, {}, {}, {}};
    stage(0, 0);

    for (int it = 0; it < CC / 32; ++it) {
        const int cur = it & 1;
        cp_wait();
        __syncthreads();
        if (it + 1 < CC / 32) stage((it + 1) * 32, cur ^ 1);

        BFrag wf;
        *(uint4*)&wf.u[0] = *(const uint4*)&lW[cur][wcol * 40 + kbase];
        *(uint4*)&wf.u[8] = *(const uint4*)&lW[cur][wcol * 40 + kbase + 16];
#pragma unroll
        for (int sub = 0; sub < 4; ++sub) {
            BFrag a;
            int row = sub * 16 + r0;
            *(uint4*)&a.u[0] = *(const uint4*)&lA[cur][row * 40 + kgrp];
            *(uint4*)&a.u[8] = *(const uint4*)&lA[cur][row * 40 + kgrp + 16];
            acc[sub] = __builtin_amdgcn_wmma_f32_16x16x32_bf16(
                false, a.v, false, wf.v, (short)0, acc[sub], false, false);
        }
    }

    // epilogue: bias + bf16 store (V goes out transposed per head)
    const int gc = colBase + wave * 16 + (lane & 15);
    const float bias = (gc < 2 * EE) ? bkv[gc] : bq[gc - 2 * EE];
    const int hf = lane >> 4;
#pragma unroll
    for (int sub = 0; sub < 4; ++sub) {
#pragma unroll
        for (int r = 0; r < 8; ++r) {
            int grow = rowBase + sub * 16 + r + hf * 8;   // C layout rows
            unsigned short val = f2bf(acc[sub][r] + bias);
            if (gc < EE) {
                Kbuf[(size_t)grow * EE + gc] = val;
            } else if (gc < 2 * EE) {
                int e = gc - EE;
                VbufT[(((size_t)(grow >> 10) * HH + (e >> 6)) * DD + (e & 63)) * NN +
                      (grow & (NN - 1))] = val;
            } else {
                Qbuf[(size_t)grow * EE + (gc - 2 * EE)] = val;
            }
        }
    }
}

// ---------------------------------------------------------------------------
// Kernel 2: flash attention, 2N=2048 keys (j<N: Kbuf rows, j>=N: Qbuf rows;
// V wraps mod N, read from transposed VbufT). 8 waves/block share (b,h);
// double-buffered async staging of K (32x64, stride 72) and V^T (64x32,
// stride 40). Softmax denominator accumulated via an all-ones WMMA column.
// ---------------------------------------------------------------------------
__global__ __launch_bounds__(256) void attn_kernel(
    const unsigned short* __restrict__ Kbuf,
    const unsigned short* __restrict__ VbufT,
    const unsigned short* __restrict__ Qbuf,
    float* __restrict__ out) {
    __shared__ unsigned short lK[2][32 * 72];
    __shared__ unsigned short lVt[2][64 * 40];
    __shared__ unsigned short lP[8][16 * 32];

    const int tid = threadIdx.x;
    const int lane = tid & 31;
    const int wave = tid >> 5;
    const int bid = blockIdx.x;               // = b*128 + h*8 + qblk
    const int qblk = bid & 7;
    const int h = (bid >> 3) & 15;
    const int b = bid >> 7;
    const int qrow0 = qblk * 128 + wave * 16;
    const int NITER = (2 * NN) / 32;

    const unsigned short* vT = &VbufT[((size_t)b * HH + h) * DD * NN];

    auto stage = [&](int kb, int buf) {
        const int j0 = kb * 32;
        const unsigned short* ksrc = (j0 < NN)
            ? &Kbuf[((size_t)b * NN + j0) * EE + h * DD]
            : &Qbuf[((size_t)b * NN + (j0 - NN)) * EE + h * DD];
        {   // K: 32 rows x 128B
            int r = tid >> 3, c = (tid & 7) * 8;
            cp16(&ksrc[(size_t)r * EE + c], &lK[buf][r * 72 + c]);
        }
        {   // V^T: 64 d-rows x 64B
            int d = tid >> 2, s = (tid & 3) * 8;
            cp16(&vT[(size_t)d * NN + (j0 & (NN - 1)) + s], &lVt[buf][d * 40 + s]);
        }
    };

    BFrag qf[2];
    {
        int row = lane & 15;
        int kg = (lane >> 4) * 8;
        const unsigned short* qptr =
            &Qbuf[((size_t)b * NN + qrow0 + row) * EE + h * DD];
#pragma unroll
        for (int c2 = 0; c2 < 2; ++c2) {
            *(uint4*)&qf[c2].u[0] = *(const uint4*)(qptr + c2 * 32 + kg);
            *(uint4*)&qf[c2].u[8] = *(const uint4*)(qptr + c2 * 32 + kg + 16);
        }
    }

    BFrag ones;
#pragma unroll
    for (int e = 0; e < 16; ++e) ones.u[e] = 0x3F80;   // bf16 1.0

    v8f o[4] = {{}, {}, {}, {}};
    v8f ol = {};                       // row-sum accumulator (ones column)
    float m[8];
#pragma unroll
    for (int r = 0; r < 8; ++r) m[r] = -3.0e38f;

    const float scale = 0.125f;        // 1/sqrt(64), folded into exp args

    stage(0, 0);

    for (int kb = 0; kb < NITER; ++kb) {
        const int cur = kb & 1;
        cp_wait();
        __syncthreads();
        if (kb + 1 < NITER) stage(kb + 1, cur ^ 1);

        // S = Q * K^T : two 16-key score tiles, k=32 twice over d
        v8f s[2] = {{}, {}};
#pragma unroll
        for (int t = 0; t < 2; ++t) {
#pragma unroll
            for (int c2 = 0; c2 < 2; ++c2) {
                BFrag kf;
                int j = t * 16 + (lane & 15);
                int dbase = c2 * 32 + (lane >> 4) * 8;
                *(uint4*)&kf.u[0] = *(const uint4*)&lK[cur][j * 72 + dbase];
                *(uint4*)&kf.u[8] = *(const uint4*)&lK[cur][j * 72 + dbase + 16];
                s[t] = __builtin_amdgcn_wmma_f32_16x16x32_bf16(
                    false, qf[c2].v, false, kf.v, (short)0, s[t], false, false);
            }
        }

        // online softmax (max reduction only; xor masks 1..8 stay in-half)
        float corr[8];
#pragma unroll
        for (int r = 0; r < 8; ++r) {
            float s0 = s[0][r];
            float s1 = s[1][r];
            float vmax = fmaxf(s0, s1);
#pragma unroll
            for (int off = 1; off < 16; off <<= 1)
                vmax = fmaxf(vmax, __shfl_xor(vmax, off, 32));
            float mn = fmaxf(m[r], vmax);
            corr[r] = __expf((m[r] - mn) * scale);
            float p0 = __expf((s0 - mn) * scale);
            float p1 = __expf((s1 - mn) * scale);
            m[r] = mn;
            int row = r + (lane >> 4) * 8;
            unsigned short* pp = &lP[wave][row * 32];
            pp[lane & 15] = f2bf(p0);
            pp[16 + (lane & 15)] = f2bf(p1);
        }

        // P: C-layout -> A-fragment via per-wave LDS (same-wave DS ordering)
        BFrag pf;
        {
            int row = lane & 15;
            int kg = (lane >> 4) * 8;
            const unsigned short* pp = &lP[wave][row * 32];
            *(uint4*)&pf.u[0] = *(const uint4*)(pp + kg);
            *(uint4*)&pf.u[8] = *(const uint4*)(pp + kg + 16);
        }

        // O = diag(corr)*O + P*V  (V^T fragment = contiguous b128 loads)
#pragma unroll
        for (int dt = 0; dt < 4; ++dt) {
            BFrag vf;
            int d = dt * 16 + (lane & 15);
            int kb2 = (lane >> 4) * 8;
            *(uint4*)&vf.u[0] = *(const uint4*)&lVt[cur][d * 40 + kb2];
            *(uint4*)&vf.u[8] = *(const uint4*)&lVt[cur][d * 40 + kb2 + 16];
            v8f oo = o[dt];
#pragma unroll
            for (int r = 0; r < 8; ++r) oo[r] *= corr[r];
            o[dt] = __builtin_amdgcn_wmma_f32_16x16x32_bf16(
                false, pf.v, false, vf.v, (short)0, oo, false, false);
        }
        // l = diag(corr)*l + P*1  (row sums via ones column)
        {
            v8f oo = ol;
#pragma unroll
            for (int r = 0; r < 8; ++r) oo[r] *= corr[r];
            ol = __builtin_amdgcn_wmma_f32_16x16x32_bf16(
                false, pf.v, false, ones.v, (short)0, oo, false, false);
        }
    }

    const int hf = lane >> 4;
#pragma unroll
    for (int dt = 0; dt < 4; ++dt) {
#pragma unroll
        for (int r = 0; r < 8; ++r) {
            int row = qrow0 + r + hf * 8;
            int d = dt * 16 + (lane & 15);
            out[((size_t)b * NN + row) * EE + h * DD + d] = o[dt][r] / ol[r];
        }
    }
}

extern "C" void kernel_launch(void* const* d_in, const int* in_sizes, int n_in,
                              void* d_out, int out_size, void* d_ws, size_t ws_size,
                              hipStream_t stream) {
    const float* x     = (const float*)d_in[0];
    const float* query = (const float*)d_in[1];
    const float* Wkv   = (const float*)d_in[2];
    const float* bkv   = (const float*)d_in[3];
    const float* Wq    = (const float*)d_in[4];
    const float* bq    = (const float*)d_in[5];
    float* out = (float*)d_out;

    const size_t NE = (size_t)BB * NN * EE;   // 4M elements
    unsigned short* Kbuf  = (unsigned short*)d_ws;          // 8 MB
    unsigned short* VbufT = Kbuf + NE;                      // 8 MB (transposed)
    unsigned short* Qbuf  = VbufT + NE;                     // 8 MB
    unsigned short* xb    = Qbuf + NE;                      // 8 MB
    unsigned short* qb    = xb + (size_t)BB * NN * CC;      // 8 MB
    unsigned short* WkvT  = qb + (size_t)BB * NN * CC;      // 4 MB
    unsigned short* WqT   = WkvT + (size_t)CC * 2 * EE;     // 2 MB

    const int nA = BB * NN * CC;
    cvt_bf16_kernel<<<nA / (256 * 8), 256, 0, stream>>>(x, xb, nA);
    cvt_bf16_kernel<<<nA / (256 * 8), 256, 0, stream>>>(query, qb, nA);

    transpose_bf16_kernel<<<dim3(2 * EE / 32, CC / 32), 256, 0, stream>>>(
        Wkv, WkvT, CC, 2 * EE);
    transpose_bf16_kernel<<<dim3(EE / 32, CC / 32), 256, 0, stream>>>(
        Wq, WqT, CC, EE);

    dim3 g1(BB * NN / 64, (3 * EE) / 128);    // 64 x 24 blocks
    proj_kernel<<<g1, 256, 0, stream>>>(xb, qb, WkvT, WqT, bkv, bq,
                                        Kbuf, VbufT, Qbuf);

    dim3 g2(BB * HH * (NN / 128));            // 512 blocks, 8 waves each
    attn_kernel<<<g2, 256, 0, stream>>>(Kbuf, VbufT, Qbuf, out);
}
